// Attention_msa_86517821215096
// MI455X (gfx1250) — matrix operand order repaired and checked by reading, not compiled, not run on hardware
//
#include <hip/hip_runtime.h>

// ---------------------------------------------------------------------------
// CDNA5 (gfx1250) wave32 WMMA attention pipeline.
// ---------------------------------------------------------------------------

typedef __attribute__((ext_vector_type(16))) _Float16 v16h;
typedef __attribute__((ext_vector_type(8)))  float    v8f;
typedef __attribute__((ext_vector_type(4)))  float    f32x4;

#define DEV __device__ __forceinline__

static constexpr int NN = 2048;   // tokens
static constexpr int CC = 1024;   // channels
static constexpr int HH = 8;      // heads
static constexpr int DD = 128;    // head dim
static constexpr int NC = 3072;   // 3*C

DEV v8f wmma_f16(v16h a, v16h b, v8f c) {
  // D = A(16x32 f16) * B(32x16 f16) + C(16x16 f32)
  return __builtin_amdgcn_wmma_f32_16x16x32_f16(
      false, a, false, b, (short)0, c, false, false);
}

// A-fragment: 16x32 f16 tile at (r0, k0) of row-major [rows x ld].
// ISA layout: lane m = lane&15; K halves [kb, kb+8) and [kb+16, kb+24),
// kb = (lane>>4)*8.  Requires ld, k0 multiples of 8 (16B aligned chunks).
DEV v16h load_frag_a(const _Float16* p, int ld, int r0, int k0) {
  int lane = threadIdx.x & 31;
  const _Float16* base =
      p + (size_t)(r0 + (lane & 15)) * ld + k0 + ((lane >> 4) << 3);
  v16h f;
  ((f32x4*)&f)[0] = *(const f32x4*)(base);
  ((f32x4*)&f)[1] = *(const f32x4*)(base + 16);
  return f;
}

// B-fragment: 32x16 f16 B-matrix, sourced from row-major B^T [N x K] buffer.
// ISA layout: lane n = lane&15; K halves [kb, kb+8) and [kb+16, kb+24),
// kb = (lane>>4)*8 (same K order as the A-fragment).
DEV v16h load_frag_b(const _Float16* p, int ld, int n0, int k0) {
  int lane = threadIdx.x & 31;
  const _Float16* base =
      p + (size_t)(n0 + (lane & 15)) * ld + k0 + ((lane >> 4) << 3);
  v16h f;
  ((f32x4*)&f)[0] = *(const f32x4*)(base);
  ((f32x4*)&f)[1] = *(const f32x4*)(base + 16);
  return f;
}

DEV float block_reduce_max(float v, float* red) {
  int t = threadIdx.x;
  red[t] = v; __syncthreads();
  for (int s = 128; s > 0; s >>= 1) {
    if (t < s) red[t] = fmaxf(red[t], red[t + s]);
    __syncthreads();
  }
  float r = red[0]; __syncthreads();
  return r;
}

DEV float block_reduce_sum(float v, float* red) {
  int t = threadIdx.x;
  red[t] = v; __syncthreads();
  for (int s = 128; s > 0; s >>= 1) {
    if (t < s) red[t] += red[t + s];
    __syncthreads();
  }
  float r = red[0]; __syncthreads();
  return r;
}

// ---------------------------------------------------------------------------
// K1: f32 -> f16 elementwise
__global__ void __launch_bounds__(256)
k_f32_to_f16(const float* __restrict__ in, _Float16* __restrict__ out, int n) {
  int i = blockIdx.x * 256 + threadIdx.x;
  if (i < n) out[i] = (_Float16)in[i];
}

// K2: W [K x N] f32 -> W^T [N x K] f16
__global__ void __launch_bounds__(256)
k_transpose_f16(const float* __restrict__ in, _Float16* __restrict__ out,
                int K, int N) {
  int i = blockIdx.x * 256 + threadIdx.x;
  if (i >= K * N) return;
  int n = i / K, k = i - n * K;
  out[i] = (_Float16)in[(size_t)k * N + n];
}

// ---------------------------------------------------------------------------
// K3: C[M x N] f32 = A16[M x K] @ (Bt16[N x K])^T   (qkv projection)
// 256 threads = 8 waves (4 m-rows x 2 n-cols); wave tile 32x64; block 128x128.
__global__ void __launch_bounds__(256)
k_gemm(const _Float16* __restrict__ A, const _Float16* __restrict__ Bt,
       float* __restrict__ C, int M, int N, int K) {
  int wave = threadIdx.x >> 5;
  int m0 = blockIdx.y * 128 + (wave & 3) * 32;
  int n0 = blockIdx.x * 128 + (wave >> 2) * 64;
  v8f acc[2][4] = {};
  for (int k0 = 0; k0 < K; k0 += 32) {
    v16h a0 = load_frag_a(A, K, m0,      k0);
    v16h a1 = load_frag_a(A, K, m0 + 16, k0);
    v16h b0 = load_frag_b(Bt, K, n0,      k0);
    v16h b1 = load_frag_b(Bt, K, n0 + 16, k0);
    v16h b2 = load_frag_b(Bt, K, n0 + 32, k0);
    v16h b3 = load_frag_b(Bt, K, n0 + 48, k0);
    acc[0][0] = wmma_f16(a0, b0, acc[0][0]);
    acc[0][1] = wmma_f16(a0, b1, acc[0][1]);
    acc[0][2] = wmma_f16(a0, b2, acc[0][2]);
    acc[0][3] = wmma_f16(a0, b3, acc[0][3]);
    acc[1][0] = wmma_f16(a1, b0, acc[1][0]);
    acc[1][1] = wmma_f16(a1, b1, acc[1][1]);
    acc[1][2] = wmma_f16(a1, b2, acc[1][2]);
    acc[1][3] = wmma_f16(a1, b3, acc[1][3]);
  }
  int lane = threadIdx.x & 31;
  int cn = lane & 15, rb = (lane >> 4) << 3;
  #pragma unroll
  for (int i = 0; i < 2; i++)
    #pragma unroll
    for (int j = 0; j < 4; j++)
      #pragma unroll
      for (int r = 0; r < 8; r++)
        C[(size_t)(m0 + 16 * i + rb + r) * N + (n0 + 16 * j + cn)] =
            acc[i][j][r];
}

// ---------------------------------------------------------------------------
// K4: per-(head,row) L2 norms + pack head-major f16 tensors + x_ori output.
// 1 wave per (h,n); 8 waves per block.
__global__ void __launch_bounds__(256)
k_normalize_pack(const float* __restrict__ qkvc, const float* __restrict__ qkvr,
                 _Float16* __restrict__ qn_c, _Float16* __restrict__ kn_c,
                 _Float16* __restrict__ vn_c, _Float16* __restrict__ vT_c,
                 _Float16* __restrict__ qn_r, _Float16* __restrict__ kn_r,
                 float* __restrict__ out_x) {
  int wave = threadIdx.x >> 5;
  int lane = threadIdx.x & 31;
  int pair = blockIdx.x * 8 + wave;      // h*2048 + n
  int h = pair >> 11;
  int n = pair & (NN - 1);
  const float* rowc = qkvc + (size_t)n * NC;
  const float* rowr = qkvr + (size_t)n * NC;
  float qc[4], kc[4], vc[4], qr[4], kr[4];
  float sq = 0.f, sk = 0.f, sv = 0.f, sqr = 0.f, skr = 0.f;
  #pragma unroll
  for (int i = 0; i < 4; i++) {
    int d = lane + 32 * i;
    qc[i] = rowc[h * DD + d];
    kc[i] = rowc[CC + h * DD + d];
    vc[i] = rowc[2 * CC + h * DD + d];
    qr[i] = rowr[h * DD + d];
    kr[i] = rowr[CC + h * DD + d];
    sq += qc[i] * qc[i]; sk += kc[i] * kc[i]; sv += vc[i] * vc[i];
    sqr += qr[i] * qr[i]; skr += kr[i] * kr[i];
  }
  #pragma unroll
  for (int m = 16; m >= 1; m >>= 1) {
    sq  += __shfl_xor(sq,  m, 32);
    sk  += __shfl_xor(sk,  m, 32);
    sv  += __shfl_xor(sv,  m, 32);
    sqr += __shfl_xor(sqr, m, 32);
    skr += __shfl_xor(skr, m, 32);
  }
  float iq = 1.f / sqrtf(sq), ik = 1.f / sqrtf(sk), iv = 1.f / sqrtf(sv);
  float iqr = 1.f / sqrtf(sqr), ikr = 1.f / sqrtf(skr);
  size_t base = ((size_t)h * NN + n) * DD;
  #pragma unroll
  for (int i = 0; i < 4; i++) {
    int d = lane + 32 * i;
    qn_c[base + d] = (_Float16)(qc[i] * iq);
    kn_c[base + d] = (_Float16)(kc[i] * ik);
    vn_c[base + d] = (_Float16)(vc[i] * iv);
    qn_r[base + d] = (_Float16)(qr[i] * iqr);
    kn_r[base + d] = (_Float16)(kr[i] * ikr);
    vT_c[((size_t)h * DD + d) * NN + n] = (_Float16)vc[i];   // unnormalized v^T
    out_x[(size_t)n * (2 * CC) + CC + h * DD + d] = vc[i];   // x_ori (exact f32)
  }
}

// ---------------------------------------------------------------------------
// K5: per head: logits_cls, logits_reg, vn@vn^T (raw similarity).
// 8 waves (2 m-rows x 4 n-cols); wave tile 32x32; block 64x128; K=128.
__global__ void __launch_bounds__(256)
k_logits(const _Float16* __restrict__ qc, const _Float16* __restrict__ kc,
         const _Float16* __restrict__ qr, const _Float16* __restrict__ kr,
         const _Float16* __restrict__ vn,
         const float* __restrict__ cls_score, const float* __restrict__ fg_score,
         float* __restrict__ Lc, float* __restrict__ Lr,
         float* __restrict__ rawmean, int head) {
  int wave = threadIdx.x >> 5;
  int m0 = blockIdx.y * 64 + (wave >> 2) * 32;
  int n0 = blockIdx.x * 128 + (wave & 3) * 32;
  v8f aC[2][2] = {}, aR[2][2] = {}, aV[2][2] = {};
  for (int k0 = 0; k0 < DD; k0 += 32) {
    v16h a0 = load_frag_a(qc, DD, m0, k0);
    v16h a1 = load_frag_a(qc, DD, m0 + 16, k0);
    v16h b0 = load_frag_b(kc, DD, n0, k0);
    v16h b1 = load_frag_b(kc, DD, n0 + 16, k0);
    aC[0][0] = wmma_f16(a0, b0, aC[0][0]);
    aC[0][1] = wmma_f16(a0, b1, aC[0][1]);
    aC[1][0] = wmma_f16(a1, b0, aC[1][0]);
    aC[1][1] = wmma_f16(a1, b1, aC[1][1]);
    a0 = load_frag_a(qr, DD, m0, k0);
    a1 = load_frag_a(qr, DD, m0 + 16, k0);
    b0 = load_frag_b(kr, DD, n0, k0);
    b1 = load_frag_b(kr, DD, n0 + 16, k0);
    aR[0][0] = wmma_f16(a0, b0, aR[0][0]);
    aR[0][1] = wmma_f16(a0, b1, aR[0][1]);
    aR[1][0] = wmma_f16(a1, b0, aR[1][0]);
    aR[1][1] = wmma_f16(a1, b1, aR[1][1]);
    a0 = load_frag_a(vn, DD, m0, k0);
    a1 = load_frag_a(vn, DD, m0 + 16, k0);
    b0 = load_frag_b(vn, DD, n0, k0);
    b1 = load_frag_b(vn, DD, n0 + 16, k0);
    aV[0][0] = wmma_f16(a0, b0, aV[0][0]);
    aV[0][1] = wmma_f16(a0, b1, aV[0][1]);
    aV[1][0] = wmma_f16(a1, b0, aV[1][0]);
    aV[1][1] = wmma_f16(a1, b1, aV[1][1]);
  }
  int lane = threadIdx.x & 31;
  #pragma unroll
  for (int ti = 0; ti < 2; ti++) {
    #pragma unroll
    for (int tj = 0; tj < 2; tj++) {
      int col  = n0 + 16 * tj + (lane & 15);
      int rowb = m0 + 16 * ti + ((lane >> 4) << 3);
      float sc_c = cls_score[col];   // cs_col: score of key (column) index
      float sc_r = fg_score[col];
      #pragma unroll
      for (int r = 0; r < 8; r++) {
        int row = rowb + r;
        float mc = (sc_c > cls_score[row] - 0.1f) ? 1.f : 0.f;
        float mr = (sc_r > fg_score[row] - 0.1f) ? 1.f : 0.f;
        size_t idx = (size_t)row * NN + col;
        Lc[idx] = aC[ti][tj][r] * 25.0f * sc_c * mc;
        Lr[idx] = aR[ti][tj][r] * 25.0f * sc_r * mr;
        float rv = aV[ti][tj][r] * 0.125f;
        rawmean[idx] = (head == 0) ? rv : rawmean[idx] + rv;
      }
    }
  }
}

// ---------------------------------------------------------------------------
// K6: per head: row softmax of Lc and Lr, attn = avg, accumulate sim_attn.
__global__ void __launch_bounds__(256)
k_softmax_combine(const float* __restrict__ Lc, const float* __restrict__ Lr,
                  _Float16* __restrict__ attn16, float* __restrict__ simattn,
                  int head) {
  __shared__ float sc[NN];
  __shared__ float sr[NN];
  __shared__ float red[256];
  int n = blockIdx.x, t = threadIdx.x;
  const float* lc = Lc + (size_t)n * NN;
  const float* lr = Lr + (size_t)n * NN;
  float mc = -1e30f, mr = -1e30f;
  for (int j = t; j < NN; j += 256) {
    float a = lc[j]; sc[j] = a; mc = fmaxf(mc, a);
    float b = lr[j]; sr[j] = b; mr = fmaxf(mr, b);
  }
  __syncthreads();
  mc = block_reduce_max(mc, red);
  mr = block_reduce_max(mr, red);
  float sumc = 0.f, sumr = 0.f;
  for (int j = t; j < NN; j += 256) {
    float ec = expf(sc[j] - mc); sc[j] = ec; sumc += ec;
    float er = expf(sr[j] - mr); sr[j] = er; sumr += er;
  }
  __syncthreads();
  sumc = block_reduce_sum(sumc, red);
  sumr = block_reduce_sum(sumr, red);
  float ic = 1.f / sumc, ir = 1.f / sumr;
  for (int j = t; j < NN; j += 256) {
    float a = 0.5f * (sc[j] * ic + sr[j] * ir);
    size_t idx = (size_t)n * NN + j;
    attn16[idx] = (_Float16)a;
    simattn[idx] = (head == 0) ? a * 0.125f : simattn[idx] + a * 0.125f;
  }
}

// ---------------------------------------------------------------------------
// K7: per head: x[:, h*128:(h+1)*128] = attn @ v  (M=2048, N=128, K=2048).
// 8 waves (4 m-rows x 2 n-cols); wave tile 32x64; block 128x128; grid (1,16).
__global__ void __launch_bounds__(256)
k_av(const _Float16* __restrict__ attn16, const _Float16* __restrict__ vT,
     float* __restrict__ out_x, int head) {
  int wave = threadIdx.x >> 5;
  int m0 = blockIdx.y * 128 + (wave & 3) * 32;
  int n0 = (wave >> 2) * 64;
  v8f acc[2][4] = {};
  for (int k0 = 0; k0 < NN; k0 += 32) {
    v16h a0 = load_frag_a(attn16, NN, m0,      k0);
    v16h a1 = load_frag_a(attn16, NN, m0 + 16, k0);
    v16h b0 = load_frag_b(vT, NN, n0,      k0);
    v16h b1 = load_frag_b(vT, NN, n0 + 16, k0);
    v16h b2 = load_frag_b(vT, NN, n0 + 32, k0);
    v16h b3 = load_frag_b(vT, NN, n0 + 48, k0);
    acc[0][0] = wmma_f16(a0, b0, acc[0][0]);
    acc[0][1] = wmma_f16(a0, b1, acc[0][1]);
    acc[0][2] = wmma_f16(a0, b2, acc[0][2]);
    acc[0][3] = wmma_f16(a0, b3, acc[0][3]);
    acc[1][0] = wmma_f16(a1, b0, acc[1][0]);
    acc[1][1] = wmma_f16(a1, b1, acc[1][1]);
    acc[1][2] = wmma_f16(a1, b2, acc[1][2]);
    acc[1][3] = wmma_f16(a1, b3, acc[1][3]);
  }
  int lane = threadIdx.x & 31;
  int cn = lane & 15, rb = (lane >> 4) << 3;
  #pragma unroll
  for (int i = 0; i < 2; i++)
    #pragma unroll
    for (int j = 0; j < 4; j++)
      #pragma unroll
      for (int r = 0; r < 8; r++)
        out_x[(size_t)(m0 + 16 * i + rb + r) * (2 * CC) +
              head * DD + n0 + 16 * j + cn] = acc[i][j][r];
}

// ---------------------------------------------------------------------------
// K8: sim_round2 = renorm( (rawmean > 0.75) * softmax(simattn) )
__global__ void __launch_bounds__(256)
k_sim_final(const float* __restrict__ simattn, const float* __restrict__ rawmean,
            float* __restrict__ out_sim) {
  __shared__ float s[NN];
  __shared__ float red[256];
  int n = blockIdx.x, t = threadIdx.x;
  const float* row = simattn + (size_t)n * NN;
  const float* rm  = rawmean + (size_t)n * NN;
  float mx = -1e30f;
  for (int j = t; j < NN; j += 256) { float a = row[j]; s[j] = a; mx = fmaxf(mx, a); }
  __syncthreads();
  mx = block_reduce_max(mx, red);
  float sum = 0.f;
  for (int j = t; j < NN; j += 256) { float e = expf(s[j] - mx); s[j] = e; sum += e; }
  __syncthreads();
  sum = block_reduce_sum(sum, red);
  float inv = 1.f / sum;
  float tot = 0.f;
  for (int j = t; j < NN; j += 256) {
    float v = (rm[j] > 0.75f) ? s[j] * inv : 0.f;
    s[j] = v; tot += v;
  }
  __syncthreads();
  tot = block_reduce_sum(tot, red);
  float itot = 1.f / tot;
  for (int j = t; j < NN; j += 256)
    out_sim[(size_t)n * NN + j] = s[j] * itot;
}

// ---------------------------------------------------------------------------
extern "C" void kernel_launch(void* const* d_in, const int* in_sizes, int n_in,
                              void* d_out, int out_size, void* d_ws, size_t ws_size,
                              hipStream_t stream) {
  (void)in_sizes; (void)n_in; (void)out_size; (void)ws_size;
  const float* x_cls     = (const float*)d_in[0];
  const float* x_reg     = (const float*)d_in[1];
  const float* cls_score = (const float*)d_in[2];
  const float* fg_score  = (const float*)d_in[3];
  const float* W_cls     = (const float*)d_in[4];
  const float* W_reg     = (const float*)d_in[5];

  float* out_x   = (float*)d_out;                     // (1, 2048, 2048)
  float* out_sim = out_x + (size_t)NN * 2 * CC;       // (2048, 2048)

  // ---- workspace layout (all offsets 256B aligned, total ~124 MB) ----
  char* ws = (char*)d_ws;
  size_t off = 0;
  auto take = [&](size_t bytes) {
    char* p = ws + off;
    off = (off + bytes + 255) & ~(size_t)255;
    return p;
  };
  _Float16* x16c  = (_Float16*)take((size_t)NN * CC * 2);   // dead after GEMM
  _Float16* x16r  = (_Float16*)take((size_t)NN * CC * 2);   // dead after GEMM
  _Float16* wT16c = (_Float16*)take((size_t)NC * CC * 2);
  _Float16* wT16r = (_Float16*)take((size_t)NC * CC * 2);
  float*    qkvc  = (float*)take((size_t)NN * NC * 4);      // dead after pack
  float*    qkvr  = (float*)take((size_t)NN * NC * 4);      // dead after pack
  _Float16* qn_c  = (_Float16*)take((size_t)HH * NN * DD * 2);
  _Float16* kn_c  = (_Float16*)take((size_t)HH * NN * DD * 2);
  _Float16* vn_c  = (_Float16*)take((size_t)HH * NN * DD * 2);
  _Float16* vT_c  = (_Float16*)take((size_t)HH * DD * NN * 2);
  _Float16* qn_r  = (_Float16*)take((size_t)HH * NN * DD * 2);
  _Float16* kn_r  = (_Float16*)take((size_t)HH * NN * DD * 2);
  float*    rawmean = (float*)take((size_t)NN * NN * 4);
  float*    simattn = (float*)take((size_t)NN * NN * 4);
  // reuse dead regions (sequential stream ordering guarantees safety):
  float*    Lc     = qkvc;                 // 16.8 MB into 25.2 MB region
  float*    Lr     = qkvr;
  _Float16* attn16 = x16c;                 // 8.4 MB spans x16c+x16r

  // ---- 1) precision conversion ----
  k_f32_to_f16<<<(NN * CC + 255) / 256, 256, 0, stream>>>(x_cls, x16c, NN * CC);
  k_f32_to_f16<<<(NN * CC + 255) / 256, 256, 0, stream>>>(x_reg, x16r, NN * CC);
  k_transpose_f16<<<(CC * NC + 255) / 256, 256, 0, stream>>>(W_cls, wT16c, CC, NC);
  k_transpose_f16<<<(CC * NC + 255) / 256, 256, 0, stream>>>(W_reg, wT16r, CC, NC);

  // ---- 2) QKV projections (WMMA) ----
  dim3 gqkv(NC / 128, NN / 128);
  k_gemm<<<gqkv, 256, 0, stream>>>(x16c, wT16c, qkvc, NN, NC, CC);
  k_gemm<<<gqkv, 256, 0, stream>>>(x16r, wT16r, qkvr, NN, NC, CC);

  // ---- 3) L2-normalize + head-major pack + x_ori ----
  k_normalize_pack<<<(HH * NN) / 8, 256, 0, stream>>>(
      qkvc, qkvr, qn_c, kn_c, vn_c, vT_c, qn_r, kn_r, out_x);

  // ---- 4) per-head attention ----
  for (int h = 0; h < HH; h++) {
    size_t ho = (size_t)h * NN * DD;
    k_logits<<<dim3(NN / 128, NN / 64), 256, 0, stream>>>(
        qn_c + ho, kn_c + ho, qn_r + ho, kn_r + ho, vn_c + ho,
        cls_score, fg_score, Lc, Lr, rawmean, h);
    k_softmax_combine<<<NN, 256, 0, stream>>>(Lc, Lr, attn16, simattn, h);
    k_av<<<dim3(1, NN / 128), 256, 0, stream>>>(
        attn16, vT_c + (size_t)h * DD * NN, out_x, h);
  }

  // ---- 5) sim_round2 ----
  k_sim_final<<<NN, 256, 0, stream>>>(simattn, rawmean, out_sim);
}
